// AttentionHead_6502580486211
// MI455X (gfx1250) — matrix operand order repaired and checked by reading, not compiled, not run on hardware
//
#include <hip/hip_runtime.h>
#include <hip/hip_bf16.h>
#include <stdint.h>

// Problem constants (match reference)
#define BB   4
#define TT   2048
#define CC   1024
#define NHH  16
#define HDD  64
#define M_ROWS (BB*TT)   // 8192
#define N_QKV  (3*CC)    // 3072

typedef __attribute__((ext_vector_type(8)))  float  v8f;
typedef __attribute__((ext_vector_type(16))) __bf16 v16bf;

__device__ __forceinline__ unsigned short f32_to_bf16_rne(float f) {
  unsigned int u = __builtin_bit_cast(unsigned int, f);
  u += 0x7FFFu + ((u >> 16) & 1u);
  return (unsigned short)(u >> 16);
}

union FragU { v16bf v; uint4 u[2]; };

// A-matrix 16x32 bf16 fragment (this code's contiguous k order, matched by B): lane m = lane&15 selects row,
// halves e=0..15 hold K = base+e, base=(lane&16)?16:0.
// => two 16-byte loads per lane from a row-major source.
__device__ __forceinline__ v16bf load_a_frag(const unsigned short* base, int ldh,
                                             int row0, int kb, int lane) {
  int m  = lane & 15;
  int k2 = kb + ((lane & 16) ? 16 : 0);
  const unsigned short* p = base + (size_t)(row0 + m) * ldh + k2;
  FragU f;
  f.u[0] = *(const uint4*)(p);
  f.u[1] = *(const uint4*)(p + 8);
  return f.v;
}

// B-matrix 32x16 bf16 fragment: lane n = lane&15 selects column,
// halves e=0..15 hold K = ((lane&16)?16:0)+e. With a K-contiguous source
// (transposed weights / K rows / transposed V) => two 16-byte loads.
__device__ __forceinline__ v16bf load_b_contig(const unsigned short* p) {
  FragU f;
  f.u[0] = *(const uint4*)(p);
  f.u[1] = *(const uint4*)(p + 8);
  return f.v;
}

// Row reductions for the C/D layout: a row lives across a 16-lane half-wave.
__device__ __forceinline__ float red16_max(float x) {
#pragma unroll
  for (int m = 1; m < 16; m <<= 1) x = fmaxf(x, __shfl_xor(x, m, 32));
  return x;
}
__device__ __forceinline__ float red16_sum(float x) {
#pragma unroll
  for (int m = 1; m < 16; m <<= 1) x += __shfl_xor(x, m, 32);
  return x;
}

#define WMMA_BF16(a, b, c) \
  __builtin_amdgcn_wmma_f32_16x16x32_bf16(false, (a), false, (b), (short)0, (c), false, false)

// ---------------------------------------------------------------------------
// 1a) fp32 -> bf16 conversion (for x)
// ---------------------------------------------------------------------------
__global__ void cvt_bf16_kernel(const float* __restrict__ src,
                                unsigned short* __restrict__ dst, int n) {
  int i = blockIdx.x * blockDim.x + threadIdx.x;
  int stride = gridDim.x * blockDim.x;
  for (; i < n; i += stride) dst[i] = f32_to_bf16_rne(src[i]);
}

// ---------------------------------------------------------------------------
// 1b) fp32 [K][N] -> bf16 transposed [N][K]  (LDS-tiled, both loads coalesced)
// ---------------------------------------------------------------------------
__global__ void __launch_bounds__(256)
transpose_cvt_kernel(const float* __restrict__ src,
                     unsigned short* __restrict__ dst, int K, int N) {
  __shared__ float tile[32][33];
  int kb = blockIdx.x * 32, nb = blockIdx.y * 32;
  int tx = threadIdx.x, ty = threadIdx.y;  // 32 x 8
#pragma unroll
  for (int i = ty; i < 32; i += 8)
    tile[i][tx] = src[(size_t)(kb + i) * N + nb + tx];
  __syncthreads();
#pragma unroll
  for (int i = ty; i < 32; i += 8)
    dst[(size_t)(nb + i) * K + kb + tx] = f32_to_bf16_rne(tile[tx][i]);
}

// ---------------------------------------------------------------------------
// 2) QKV GEMM: [8192,1024] x W^T[3072,1024] + bias -> Q,K (row) / V (transposed)
//    One wave = 16x64 output tile; every fragment is two b128 loads.
// ---------------------------------------------------------------------------
__global__ void __launch_bounds__(128)
qkv_gemm_kernel(const unsigned short* __restrict__ xb,
                const unsigned short* __restrict__ wtb,  // [N_QKV][CC]
                const float* __restrict__ bias,
                unsigned short* __restrict__ q,    // [B,NH,T,HD]
                unsigned short* __restrict__ k,    // [B,NH,T,HD]
                unsigned short* __restrict__ vt) { // [B,NH,HD,T]
  int lane = threadIdx.x & 31;
  int gw = blockIdx.x * (blockDim.x >> 5) + (threadIdx.x >> 5);
  int mt = gw / (N_QKV / 64);
  int ng = gw % (N_QKV / 64);
  int row0 = mt * 16, col0 = ng * 64;
  int kgrp = (lane & 16) ? 16 : 0;
  int n    = lane & 15;

  v8f acc0 = {}, acc1 = {}, acc2 = {}, acc3 = {};
  for (int kb = 0; kb < CC; kb += 32) {
    __builtin_prefetch(xb + (size_t)(row0 + n) * CC + kb + 128, 0, 0);
    v16bf a = load_a_frag(xb, CC, row0, kb, lane);
    const unsigned short* bp = wtb + (size_t)(col0 + n) * CC + kb + kgrp;
    acc0 = WMMA_BF16(a, load_b_contig(bp),               acc0);
    acc1 = WMMA_BF16(a, load_b_contig(bp + 16 * CC),     acc1);
    acc2 = WMMA_BF16(a, load_b_contig(bp + 32 * CC),     acc2);
    acc3 = WMMA_BF16(a, load_b_contig(bp + 48 * CC),     acc3);
  }

  int rbase = row0 + ((lane & 16) ? 8 : 0);
  v8f accs[4] = {acc0, acc1, acc2, acc3};
#pragma unroll
  for (int nt = 0; nt < 4; ++nt) {
    int col = col0 + nt * 16 + n;
    float bv = bias[col];
    int sec = col / CC;
    int cc  = col % CC;
    int h   = cc / HDD, d = cc % HDD;
    float mul = (sec == 0) ? 0.125f : 1.0f;  // HD^-0.5 = 1/8, folded into Q
#pragma unroll
    for (int i = 0; i < 8; ++i) {
      int row = rbase + i;
      int b_ = row / TT, t = row % TT;
      unsigned short val = f32_to_bf16_rne((accs[nt][i] + bv) * mul);
      if (sec == 2) {
        vt[(((size_t)b_ * NHH + h) * HDD + d) * TT + t] = val;
      } else {
        size_t idx = (((size_t)b_ * NHH + h) * TT + t) * HDD + d;
        if (sec == 0) q[idx] = val; else k[idx] = val;
      }
    }
  }
}

// ---------------------------------------------------------------------------
// 3) Causal flash attention: one wave = 16-query tile of one (b,h).
//    K (row-major) and V (pre-transposed) B-fragments load straight from
//    global as b128 pairs; LDS only for the P D-layout -> A-layout round-trip.
// ---------------------------------------------------------------------------
__global__ void __launch_bounds__(128)
attn_kernel(const unsigned short* __restrict__ q,     // [B,NH,T,HD]
            const unsigned short* __restrict__ kmat,  // [B,NH,T,HD]
            const unsigned short* __restrict__ vtm,   // [B,NH,HD,T]
            unsigned short* __restrict__ ao) {        // [B,T,C] bf16
  __shared__ unsigned short Pt_s[4 * 16 * 32];
  int lane = threadIdx.x & 31;
  int wid  = threadIdx.x >> 5;
  int g  = blockIdx.x * 4 + wid;
  int qt = g & 127;          // T/16 = 128 query tiles
  int h  = (g >> 7) & 15;
  int b_ = g >> 11;
  int q0 = qt * 16;

  const unsigned short* qbase  = q    + (((size_t)b_ * NHH + h) * TT) * HDD;
  const unsigned short* kbase  = kmat + (((size_t)b_ * NHH + h) * TT) * HDD;
  const unsigned short* vtbase = vtm  + (((size_t)b_ * NHH + h) * HDD) * TT;
  unsigned short* Pt = Pt_s + wid * 512;

  v16bf a0 = load_a_frag(qbase, HDD, q0, 0,  lane);  // Q hd 0..31
  v16bf a1 = load_a_frag(qbase, HDD, q0, 32, lane);  // Q hd 32..63

  float mrun[8], lrun[8];
#pragma unroll
  for (int i = 0; i < 8; ++i) { mrun[i] = -1e30f; lrun[i] = 0.f; }
  v8f o0 = {}, o1 = {}, o2 = {}, o3 = {};

  int nkt   = (q0 + 16 + 31) >> 5;  // 32-key tiles covering keys <= q0+15
  int kgrp  = (lane & 16) ? 16 : 0;
  int n     = lane & 15;
  int rbase = (lane & 16) ? 8 : 0;

  for (int kt = 0; kt < nkt; ++kt) {
    int k0 = kt * 32;
    __builtin_prefetch(kbase + (size_t)(k0 + 32 + lane) * HDD, 0, 0);

    // S = Q @ K^T : B fragment for key n = contiguous hd-run of K row (global)
    v8f s0 = {}, s1 = {};
    {
      const unsigned short* r0 = kbase + (size_t)(k0 + n) * HDD + kgrp;
      const unsigned short* r1 = kbase + (size_t)(k0 + 16 + n) * HDD + kgrp;
      s0 = WMMA_BF16(a0, load_b_contig(r0),      s0);
      s0 = WMMA_BF16(a1, load_b_contig(r0 + 32), s0);
      s1 = WMMA_BF16(a0, load_b_contig(r1),      s1);
      s1 = WMMA_BF16(a1, load_b_contig(r1 + 32), s1);
    }

    // causal mask needed only on the diagonal tile
    if (kt == nkt - 1) {
      int c0 = k0 + n, c1 = k0 + 16 + n;
#pragma unroll
      for (int i = 0; i < 8; ++i) {
        int row = q0 + rbase + i;
        if (c0 > row) s0[i] = -1e30f;
        if (c1 > row) s1[i] = -1e30f;
      }
    }

    // online softmax (rows live across 16-lane halves)
#pragma unroll
    for (int i = 0; i < 8; ++i) {
      float mx   = red16_max(fmaxf(s0[i], s1[i]));
      float mnew = fmaxf(mrun[i], mx);
      float corr = __expf(mrun[i] - mnew);
      float p0 = __expf(s0[i] - mnew);
      float p1 = __expf(s1[i] - mnew);
      float rs = red16_sum(p0 + p1);
      lrun[i] = lrun[i] * corr + rs;
      mrun[i] = mnew;
      o0[i] *= corr; o1[i] *= corr; o2[i] *= corr; o3[i] *= corr;
      int row = rbase + i;
      Pt[row * 32 + n]      = f32_to_bf16_rne(p0);
      Pt[row * 32 + 16 + n] = f32_to_bf16_rne(p1);
    }
    asm volatile("s_wait_dscnt 0" ::: "memory");

    // O += P @ V : P re-read in A layout from LDS; V fragment for hd column n
    // = contiguous key-run of transposed V (global)
    v16bf pA = load_a_frag(Pt, 32, 0, 0, lane);
    {
      const unsigned short* vb0 = vtbase + (size_t)n * TT + k0 + kgrp;
      o0 = WMMA_BF16(pA, load_b_contig(vb0),            o0);
      o1 = WMMA_BF16(pA, load_b_contig(vb0 + 16 * TT),  o1);
      o2 = WMMA_BF16(pA, load_b_contig(vb0 + 32 * TT),  o2);
      o3 = WMMA_BF16(pA, load_b_contig(vb0 + 48 * TT),  o3);
    }
  }

  // normalize + store attn output bf16 in [B,T,C]
  unsigned short* aob = ao + ((size_t)b_ * TT) * CC;
#pragma unroll
  for (int i = 0; i < 8; ++i) {
    float rl = 1.0f / lrun[i];
    int t = q0 + rbase + i;
    size_t base = (size_t)t * CC + h * HDD + n;
    aob[base]      = f32_to_bf16_rne(o0[i] * rl);
    aob[base + 16] = f32_to_bf16_rne(o1[i] * rl);
    aob[base + 32] = f32_to_bf16_rne(o2[i] * rl);
    aob[base + 48] = f32_to_bf16_rne(o3[i] * rl);
  }
}

// ---------------------------------------------------------------------------
// 4) Output projection: [8192,1024] x Wp^T[1024,1024] + bias -> f32 d_out
// ---------------------------------------------------------------------------
__global__ void __launch_bounds__(128)
proj_gemm_kernel(const unsigned short* __restrict__ ab,
                 const unsigned short* __restrict__ wtb,  // [CC][CC] transposed
                 const float* __restrict__ bias,
                 float* __restrict__ out) {
  int lane = threadIdx.x & 31;
  int gw = blockIdx.x * (blockDim.x >> 5) + (threadIdx.x >> 5);
  int mt = gw / (CC / 64);
  int ng = gw % (CC / 64);
  int row0 = mt * 16, col0 = ng * 64;
  int kgrp = (lane & 16) ? 16 : 0;
  int n    = lane & 15;

  v8f acc0 = {}, acc1 = {}, acc2 = {}, acc3 = {};
  for (int kb = 0; kb < CC; kb += 32) {
    __builtin_prefetch(ab + (size_t)(row0 + n) * CC + kb + 128, 0, 0);
    v16bf a = load_a_frag(ab, CC, row0, kb, lane);
    const unsigned short* bp = wtb + (size_t)(col0 + n) * CC + kb + kgrp;
    acc0 = WMMA_BF16(a, load_b_contig(bp),           acc0);
    acc1 = WMMA_BF16(a, load_b_contig(bp + 16 * CC), acc1);
    acc2 = WMMA_BF16(a, load_b_contig(bp + 32 * CC), acc2);
    acc3 = WMMA_BF16(a, load_b_contig(bp + 48 * CC), acc3);
  }

  int rb = row0 + ((lane & 16) ? 8 : 0);
  v8f accs[4] = {acc0, acc1, acc2, acc3};
#pragma unroll
  for (int nt = 0; nt < 4; ++nt) {
    int col = col0 + nt * 16 + n;
    float bv = bias[col];
#pragma unroll
    for (int i = 0; i < 8; ++i)
      out[(size_t)(rb + i) * CC + col] = accs[nt][i] + bv;
  }
}

// ---------------------------------------------------------------------------
extern "C" void kernel_launch(void* const* d_in, const int* in_sizes, int n_in,
                              void* d_out, int out_size, void* d_ws, size_t ws_size,
                              hipStream_t stream) {
  const float* x      = (const float*)d_in[0];
  const float* w_attn = (const float*)d_in[1];
  const float* b_attn = (const float*)d_in[2];
  const float* w_proj = (const float*)d_in[3];
  const float* b_proj = (const float*)d_in[4];
  float* out = (float*)d_out;

  unsigned short* ws = (unsigned short*)d_ws;
  size_t off = 0;
  unsigned short* xb   = ws + off; off += (size_t)M_ROWS * CC;  // x bf16
  unsigned short* watb = ws + off; off += (size_t)N_QKV * CC;   // w_attn^T bf16
  unsigned short* wptb = ws + off; off += (size_t)CC * CC;      // w_proj^T bf16
  unsigned short* qb   = ws + off; off += (size_t)M_ROWS * CC;  // Q [B,NH,T,HD]
  unsigned short* kbuf = ws + off; off += (size_t)M_ROWS * CC;  // K [B,NH,T,HD]
  unsigned short* vtb  = ws + off; off += (size_t)M_ROWS * CC;  // V^T [B,NH,HD,T]
  unsigned short* aob  = ws + off; off += (size_t)M_ROWS * CC;  // attn out [B,T,C]
  (void)ws_size; (void)in_sizes; (void)n_in; (void)out_size;

  cvt_bf16_kernel<<<2048, 256, 0, stream>>>(x, xb, M_ROWS * CC);
  transpose_cvt_kernel<<<dim3(CC / 32, N_QKV / 32), dim3(32, 8), 0, stream>>>(
      w_attn, watb, CC, N_QKV);
  transpose_cvt_kernel<<<dim3(CC / 32, CC / 32), dim3(32, 8), 0, stream>>>(
      w_proj, wptb, CC, CC);

  qkv_gemm_kernel<<<(M_ROWS / 16) * (N_QKV / 64) / 4, 128, 0, stream>>>(
      xb, watb, b_attn, qb, kbuf, vtb);

  attn_kernel<<<(BB * NHH * (TT / 16)) / 4, 128, 0, stream>>>(qb, kbuf, vtb, aob);

  proj_gemm_kernel<<<(M_ROWS / 16) * (CC / 64) / 4, 128, 0, stream>>>(
      aob, wptb, b_proj, out);
}
